// SwinTransformerBlock_11965778887199
// MI455X (gfx1250) — matrix operand order repaired and checked by reading, not compiled, not run on hardware
//
#include <hip/hip_runtime.h>
#include <hip/hip_fp16.h>
#include <math.h>

typedef __attribute__((ext_vector_type(16))) _Float16 v16h;
typedef __attribute__((ext_vector_type(8)))  float    v8f;
typedef _Float16 h16;

#define LDA_S 40   // padded LDS stride (halves): 80B, 16B-aligned, conflict-free for 16-row frag reads
#define LDB_S 40
#define BKK   32
#define BUFH  (128 * LDA_S)   // halves per LDS buffer

// ---- fragment loaders (contiguous k order, k = 16*(lane/16) + i) ----------------------------
// A 16x32 f16: lanes0-15 row m=lane, elems = K0..15;
//              lanes16-31 same rows, K16..31.
static __device__ __forceinline__ v16h frag_a(const h16* s, int ld) {
  const int lane = threadIdx.x & 31;
  const h16* p = s + (lane & 15) * ld + ((lane >> 4) << 4);
  v16h a;
#pragma unroll
  for (int i = 0; i < 16; ++i) a[i] = p[i];
  return a;
}
// B 32x16 f16 held transposed (as [n][k]): lanes0-15 col n=lane, elems = K0..15;
// lanes16-31 elems = K16..31.  Contiguous 16 halves per lane.
static __device__ __forceinline__ v16h frag_bt(const h16* s, int ld) {
  const int lane = threadIdx.x & 31;
  const h16* p = s + (lane & 15) * ld + ((lane >> 4) << 4);
  v16h b;
#pragma unroll
  for (int i = 0; i < 16; ++i) b[i] = p[i];
  return b;
}

// ---- shared tiled GEMM mainloop: 128x128 block tile, 8 waves (2x4), wave = 64x32
// A row-major [M][lda]; Wt = pre-transposed weights row-major [N][ldw] (ldw == K).
// Double-buffered LDS, one barrier per K-step, global loads issued ahead of WMMA.
static __device__ __forceinline__ void gemm_mainloop(
    const h16* __restrict__ A, int lda, size_t row0,
    const h16* __restrict__ Wt, int ldw, int n0, int K,
    h16* As, h16* Bs, v8f acc[4][2])
{
  const int tid = threadIdx.x;
  const int wid = tid >> 5;
  const int wm = (wid >> 2) << 6;
  const int wn = (wid & 3) << 5;
#pragma unroll
  for (int mt = 0; mt < 4; ++mt)
#pragma unroll
    for (int nt = 0; nt < 2; ++nt)
#pragma unroll
      for (int r = 0; r < 8; ++r) acc[mt][nt][r] = 0.f;

  const int r2 = tid >> 1, seg = (tid & 1) << 4;   // 128 rows x 32 halves, 2 thr/row
  const h16* aptr = A + (row0 + r2) * (size_t)lda + seg;
  const h16* bptr = Wt + (size_t)(n0 + r2) * ldw + seg;
  const int sOff = r2 * LDA_S + seg;

  uint4 ra0 = ((const uint4*)aptr)[0], ra1 = ((const uint4*)aptr)[1];
  uint4 rb0 = ((const uint4*)bptr)[0], rb1 = ((const uint4*)bptr)[1];
  *(uint4*)(As + sOff) = ra0; *(uint4*)(As + sOff + 8) = ra1;
  *(uint4*)(Bs + sOff) = rb0; *(uint4*)(Bs + sOff + 8) = rb1;
  __syncthreads();

  const int nIter = K / BKK;
  for (int i = 0; i < nIter; ++i) {
    const int buf = (i & 1) ? BUFH : 0;
    const int nbuf = buf ^ BUFH;
    if (i + 1 < nIter) {            // next K-slab: global -> regs (latency hidden under WMMA)
      const h16* ap = aptr + (i + 1) * BKK;
      const h16* bp = bptr + (i + 1) * BKK;
      ra0 = ((const uint4*)ap)[0]; ra1 = ((const uint4*)ap)[1];
      rb0 = ((const uint4*)bp)[0]; rb1 = ((const uint4*)bp)[1];
      if (i + 2 < nIter) {          // global_prefetch_b8 two slabs ahead
        __builtin_prefetch(aptr + (i + 2) * BKK, 0, 1);
        __builtin_prefetch(bptr + (i + 2) * BKK, 0, 1);
      }
    }
    v16h af[4], bf[2];
#pragma unroll
    for (int mt = 0; mt < 4; ++mt) af[mt] = frag_a(As + buf + (wm + mt * 16) * LDA_S, LDA_S);
#pragma unroll
    for (int nt = 0; nt < 2; ++nt) bf[nt] = frag_bt(Bs + buf + (wn + nt * 16) * LDB_S, LDB_S);
#pragma unroll
    for (int mt = 0; mt < 4; ++mt)
#pragma unroll
      for (int nt = 0; nt < 2; ++nt)
        acc[mt][nt] = __builtin_amdgcn_wmma_f32_16x16x32_f16(
            false, af[mt], false, bf[nt], (short)0, acc[mt][nt], false, false);
    if (i + 1 < nIter) {            // stage next slab into the other buffer
      *(uint4*)(As + nbuf + sOff) = ra0; *(uint4*)(As + nbuf + sOff + 8) = ra1;
      *(uint4*)(Bs + nbuf + sOff) = rb0; *(uint4*)(Bs + nbuf + sOff + 8) = rb1;
    }
    __syncthreads();
  }
}

// ---- weight prep: f32 [K][N] -> f16 transposed [N][K] (done once per launch) -
__global__ __launch_bounds__(256) void wt_transpose_kernel(const float* __restrict__ s,
                                                           h16* __restrict__ d, int K, int N) {
  __shared__ float t[32][33];
  const int n0 = blockIdx.x * 32, k0 = blockIdx.y * 32;
  const int tx = threadIdx.x & 31, ty = threadIdx.x >> 5;  // 32 x 8
#pragma unroll
  for (int i = 0; i < 32; i += 8)
    t[ty + i][tx] = s[(size_t)(k0 + ty + i) * N + n0 + tx];
  __syncthreads();
#pragma unroll
  for (int i = 0; i < 32; i += 8)
    d[(size_t)(n0 + ty + i) * K + k0 + tx] = (h16)t[tx][ty + i];
}

// continuous relative-position bias: tiny MLP over log-spaced coords + gather + 16*sigmoid
__global__ __launch_bounds__(256) void cpb_bias_kernel(const float* __restrict__ w1,
                                                       const float* __restrict__ b1,
                                                       const float* __restrict__ w2,
                                                       float* __restrict__ bias16) {
  __shared__ float tS[225 * 16];
  const int tid = threadIdx.x;
  if (tid < 225) {
    int xx = tid / 15, yy = tid % 15;
    float ny = (float)(yy - 7) * (1.f / 7.f);
    float nx = (float)(xx - 7) * (1.f / 7.f);
    float f0 = copysignf(log2f(fabsf(ny) * 8.f + 1.f) * (1.f / 3.f), ny);
    float f1 = copysignf(log2f(fabsf(nx) * 8.f + 1.f) * (1.f / 3.f), nx);
    float accs[16];
#pragma unroll
    for (int h = 0; h < 16; ++h) accs[h] = 0.f;
    for (int j = 0; j < 512; ++j) {
      float hj = fmaxf(f0 * w1[j] + f1 * w1[512 + j] + b1[j], 0.f);
#pragma unroll
      for (int h = 0; h < 16; ++h) accs[h] += hj * w2[j * 16 + h];
    }
#pragma unroll
    for (int h = 0; h < 16; ++h) tS[tid * 16 + h] = accs[h];
  }
  __syncthreads();
  for (int idx = tid; idx < 16 * 4096; idx += 256) {
    int h = idx >> 12, pq = idx & 4095, p = pq >> 6, q = pq & 63;
    int dx = (p >> 3) - (q >> 3), dy = (p & 7) - (q & 7);
    float tv = tS[((dx + 7) * 15 + (dy + 7)) * 16 + h];
    bias16[idx] = 16.f / (1.f + expf(-tv));
  }
}

// LN1 + cyclic shift(-4,-4) + window partition -> f16 [win][tok][512]
__global__ __launch_bounds__(256) void ln1_kernel(const float* __restrict__ x,
                                                  const float* __restrict__ g,
                                                  const float* __restrict__ b,
                                                  h16* __restrict__ hw) {
  const int win = blockIdx.x;
  const int lane = threadIdx.x & 31, wid = threadIdx.x >> 5;
  const int bi = win >> 6, w = win & 63, wi = w >> 3, wj = w & 7;
  for (int t = 0; t < 8; ++t) {
    const int tok = wid * 8 + t;
    const int gi = (wi * 8 + (tok >> 3) + 4) & 63;
    const int gj = (wj * 8 + (tok & 7) + 4) & 63;
    const float* src = x + (((size_t)bi * 64 + gi) * 64 + gj) * 512;
    float vals[16], s = 0.f, ss = 0.f;
#pragma unroll
    for (int j = 0; j < 16; ++j) { float v = src[lane + 32 * j]; vals[j] = v; s += v; ss += v * v; }
#pragma unroll
    for (int off = 16; off > 0; off >>= 1) { s += __shfl_xor(s, off, 32); ss += __shfl_xor(ss, off, 32); }
    float mean = s * (1.f / 512.f);
    float rstd = rsqrtf(ss * (1.f / 512.f) - mean * mean + 1e-5f);
    h16* dst = hw + ((size_t)win * 64 + tok) * 512;
#pragma unroll
    for (int j = 0; j < 16; ++j) {
      int c = lane + 32 * j;
      dst[c] = (h16)((vals[j] - mean) * rstd * g[c] + b[c]);
    }
  }
}

// LN2 over flat [131072][512]
__global__ __launch_bounds__(256) void ln2_kernel(const float* __restrict__ x2,
                                                  const float* __restrict__ g,
                                                  const float* __restrict__ b,
                                                  h16* __restrict__ h2) {
  const int lane = threadIdx.x & 31, wid = threadIdx.x >> 5;
  const size_t row0 = (size_t)blockIdx.x * 64 + wid * 8;
  for (int t = 0; t < 8; ++t) {
    const float* src = x2 + (row0 + t) * 512;
    float vals[16], s = 0.f, ss = 0.f;
#pragma unroll
    for (int j = 0; j < 16; ++j) { float v = src[lane + 32 * j]; vals[j] = v; s += v; ss += v * v; }
#pragma unroll
    for (int off = 16; off > 0; off >>= 1) { s += __shfl_xor(s, off, 32); ss += __shfl_xor(ss, off, 32); }
    float mean = s * (1.f / 512.f);
    float rstd = rsqrtf(ss * (1.f / 512.f) - mean * mean + 1e-5f);
    h16* dst = h2 + (row0 + t) * 512;
#pragma unroll
    for (int j = 0; j < 16; ++j) {
      int c = lane + 32 * j;
      dst[c] = (h16)((vals[j] - mean) * rstd * g[c] + b[c]);
    }
  }
}

// ---- GEMM kernels (weights pre-transposed: Wt[N][K]) ------------------------
__global__ __launch_bounds__(256) void gemm_qkv_kernel(const h16* __restrict__ A,
                                                       const h16* __restrict__ Wt,
                                                       const float* __restrict__ bias,
                                                       h16* __restrict__ q, h16* __restrict__ k,
                                                       h16* __restrict__ v) {
  __shared__ h16 As[2 * BUFH];
  __shared__ h16 Bs[2 * BUFH];
  v8f acc[4][2];
  const size_t row0 = (size_t)blockIdx.x * 128;
  const int n0 = blockIdx.y * 128;
  gemm_mainloop(A, 512, row0, Wt, 512, n0, 512, As, Bs, acc);
  const int lane = threadIdx.x & 31, wid = threadIdx.x >> 5;
  const int wm = (wid >> 2) << 6, wn = (wid & 3) << 5;
  const int hi = lane >> 4, nlo = lane & 15;
#pragma unroll
  for (int mt = 0; mt < 4; ++mt)
#pragma unroll
    for (int nt = 0; nt < 2; ++nt)
#pragma unroll
      for (int r = 0; r < 8; ++r) {
        size_t row = row0 + wm + mt * 16 + (hi << 3) + r;
        int col = n0 + wn + nt * 16 + nlo;
        float val = acc[mt][nt][r] + bias[col];
        int which = col >> 9, rest = col & 511;
        int head = rest >> 5, d = rest & 31;
        size_t win = row >> 6, tok = row & 63;
        if (which == 2) {  // V stored transposed: [win][head][d][tok] for contiguous PV frags
          v[((win * 16 + head) * 32 + d) * 64 + tok] = (h16)val;
        } else {
          h16* dst = (which == 0) ? q : k;
          dst[((win * 16 + head) * 64 + tok) * 32 + d] = (h16)val;
        }
      }
}

__global__ __launch_bounds__(256) void gemm_proj_kernel(const h16* __restrict__ A,
                                                        const h16* __restrict__ Wt,
                                                        const float* __restrict__ bias,
                                                        const float* __restrict__ xin,
                                                        float* __restrict__ xout) {
  __shared__ h16 As[2 * BUFH];
  __shared__ h16 Bs[2 * BUFH];
  v8f acc[4][2];
  const size_t row0 = (size_t)blockIdx.x * 128;
  const int n0 = blockIdx.y * 128;
  gemm_mainloop(A, 512, row0, Wt, 512, n0, 512, As, Bs, acc);
  const int lane = threadIdx.x & 31, wid = threadIdx.x >> 5;
  const int wm = (wid >> 2) << 6, wn = (wid & 3) << 5;
  const int hi = lane >> 4, nlo = lane & 15;
#pragma unroll
  for (int mt = 0; mt < 4; ++mt)
#pragma unroll
    for (int nt = 0; nt < 2; ++nt)
#pragma unroll
      for (int r = 0; r < 8; ++r) {
        size_t row = row0 + wm + mt * 16 + (hi << 3) + r;
        int col = n0 + wn + nt * 16 + nlo;
        float val = acc[mt][nt][r] + bias[col];
        size_t win = row >> 6; int tok = (int)(row & 63);
        int bb = (int)(win >> 6), w = (int)(win & 63);
        int gi = ((w >> 3) * 8 + (tok >> 3) + 4) & 63;  // window-reverse + roll(+4)
        int gj = ((w & 7) * 8 + (tok & 7) + 4) & 63;
        size_t idx = (((size_t)bb * 64 + gi) * 64 + gj) * 512 + col;
        xout[idx] = val + xin[idx];   // x2 = shortcut + attn-proj
      }
}

__global__ __launch_bounds__(256) void gemm_mlp1_kernel(const h16* __restrict__ A,
                                                        const h16* __restrict__ Wt,
                                                        const float* __restrict__ bias,
                                                        h16* __restrict__ hid) {
  __shared__ h16 As[2 * BUFH];
  __shared__ h16 Bs[2 * BUFH];
  v8f acc[4][2];
  const size_t row0 = (size_t)blockIdx.x * 128;
  const int n0 = blockIdx.y * 128;
  gemm_mainloop(A, 512, row0, Wt, 512, n0, 512, As, Bs, acc);
  const int lane = threadIdx.x & 31, wid = threadIdx.x >> 5;
  const int wm = (wid >> 2) << 6, wn = (wid & 3) << 5;
  const int hi = lane >> 4, nlo = lane & 15;
#pragma unroll
  for (int mt = 0; mt < 4; ++mt)
#pragma unroll
    for (int nt = 0; nt < 2; ++nt)
#pragma unroll
      for (int r = 0; r < 8; ++r) {
        size_t row = row0 + wm + mt * 16 + (hi << 3) + r;
        int col = n0 + wn + nt * 16 + nlo;
        float val = acc[mt][nt][r] + bias[col];
        float ge = 0.5f * val * (1.f + erff(val * 0.70710678f));  // exact GELU
        hid[row * 1024 + col] = (h16)ge;
      }
}

__global__ __launch_bounds__(256) void gemm_mlp2_kernel(const h16* __restrict__ A,
                                                        const h16* __restrict__ Wt,
                                                        const float* __restrict__ bias,
                                                        float* __restrict__ out) {
  __shared__ h16 As[2 * BUFH];
  __shared__ h16 Bs[2 * BUFH];
  v8f acc[4][2];
  const size_t row0 = (size_t)blockIdx.x * 128;
  const int n0 = blockIdx.y * 128;
  gemm_mainloop(A, 1024, row0, Wt, 1024, n0, 1024, As, Bs, acc);
  const int lane = threadIdx.x & 31, wid = threadIdx.x >> 5;
  const int wm = (wid >> 2) << 6, wn = (wid & 3) << 5;
  const int hi = lane >> 4, nlo = lane & 15;
#pragma unroll
  for (int mt = 0; mt < 4; ++mt)
#pragma unroll
    for (int nt = 0; nt < 2; ++nt)
#pragma unroll
      for (int r = 0; r < 8; ++r) {
        size_t row = row0 + wm + mt * 16 + (hi << 3) + r;
        int col = n0 + wn + nt * 16 + nlo;
        size_t idx = row * 512 + col;
        out[idx] += acc[mt][nt][r] + bias[col];   // final residual add
      }
}

// ---- fused windowed attention: one wave per (window, head, 16-row strip) ----
__global__ __launch_bounds__(256) void attn_kernel(const h16* __restrict__ Q,
                                                   const h16* __restrict__ Kx,
                                                   const h16* __restrict__ Vt,
                                                   const float* __restrict__ bias16,
                                                   const float* __restrict__ mask,
                                                   h16* __restrict__ O) {
  __shared__ h16 Pl[8][16 * 72];
  const int lane = threadIdx.x & 31, wid = threadIdx.x >> 5;
  const int gid = blockIdx.x * 8 + wid;
  const int strip = gid & 3;
  const int head = (gid >> 2) & 15;
  const int win = gid >> 6;
  const size_t base = ((size_t)win * 16 + head) * 64 * 32;
  const h16* q = Q + base;
  const h16* k = Kx + base;
  const h16* v = Vt + base;   // V transposed: [d][tok]
  const int hi = lane >> 4, nlo = lane & 15;

  // Q strip fragment (16x32, K = head-dim = 32: exactly one WMMA K-step per tile)
  v16h qa;
  {
    const h16* p = q + (size_t)(strip * 16 + nlo) * 32 + (hi << 4);
#pragma unroll
    for (int i = 0; i < 16; ++i) qa[i] = p[i];
  }
  // scores strip (16x64) = Q_strip @ K^T : 4 WMMAs
  v8f sc[4];
#pragma unroll
  for (int j = 0; j < 4; ++j) {
    v16h kb;
    const h16* p = k + (size_t)(j * 16 + nlo) * 32 + (hi << 4);
#pragma unroll
    for (int e = 0; e < 16; ++e) kb[e] = p[e];
    v8f z;
#pragma unroll
    for (int r = 0; r < 8; ++r) z[r] = 0.f;
    sc[j] = __builtin_amdgcn_wmma_f32_16x16x32_f16(false, qa, false, kb, (short)0, z, false, false);
  }

  const float* bh = bias16 + head * 4096;
  const float* mw = mask + (size_t)(win & 63) * 4096;
#pragma unroll
  for (int r = 0; r < 8; ++r) {
    const int p = strip * 16 + (hi << 3) + r;   // global query row
    float mx = -1e30f;
#pragma unroll
    for (int j = 0; j < 4; ++j) {
      const int qq = j * 16 + nlo;
      float t = sc[j][r] * 0.17677669529663689f + bh[p * 64 + qq] + mw[p * 64 + qq];
      sc[j][r] = t;
      mx = fmaxf(mx, t);
    }
    // row lives in one 16-lane half -> xor-reduce within half
#pragma unroll
    for (int off = 8; off > 0; off >>= 1) mx = fmaxf(mx, __shfl_xor(mx, off, 32));
    float sm = 0.f;
#pragma unroll
    for (int j = 0; j < 4; ++j) { float e = expf(sc[j][r] - mx); sc[j][r] = e; sm += e; }
#pragma unroll
    for (int off = 8; off > 0; off >>= 1) sm += __shfl_xor(sm, off, 32);
    float inv = 1.f / sm;
#pragma unroll
    for (int j = 0; j < 4; ++j)
      Pl[wid][((hi << 3) + r) * 72 + j * 16 + nlo] = (h16)(sc[j][r] * inv);
  }

  // O strip (16x32) = P (16x64) @ V (64x32): 2 K-steps x 2 N-tiles
  v8f ov[2];
#pragma unroll
  for (int nt = 0; nt < 2; ++nt)
#pragma unroll
    for (int r = 0; r < 8; ++r) ov[nt][r] = 0.f;
#pragma unroll
  for (int kt = 0; kt < 2; ++kt) {
    v16h pa;
    const h16* pp = &Pl[wid][nlo * 72 + kt * 32 + (hi << 4)];
#pragma unroll
    for (int i = 0; i < 16; ++i) pa[i] = pp[i];
#pragma unroll
    for (int nt = 0; nt < 2; ++nt) {
      v16h vb;  // Vt row-major [d][tok] -> contiguous 16 halves per lane
      const h16* vp = v + (size_t)(nt * 16 + nlo) * 64 + kt * 32 + (hi << 4);
#pragma unroll
      for (int e = 0; e < 16; ++e) vb[e] = vp[e];
      ov[nt] = __builtin_amdgcn_wmma_f32_16x16x32_f16(false, pa, false, vb, (short)0, ov[nt], false, false);
    }
  }
#pragma unroll
  for (int nt = 0; nt < 2; ++nt)
#pragma unroll
    for (int r = 0; r < 8; ++r) {
      int tok = strip * 16 + (hi << 3) + r;
      int d = nt * 16 + nlo;
      O[((size_t)win * 64 + tok) * 512 + head * 32 + d] = (h16)ov[nt][r];
    }
}

// ---- launch -----------------------------------------------------------------
extern "C" void kernel_launch(void* const* d_in, const int* in_sizes, int n_in,
                              void* d_out, int out_size, void* d_ws, size_t ws_size,
                              hipStream_t stream) {
  const float* x     = (const float*)d_in[0];
  const float* maskm = (const float*)d_in[1];
  const float* n1g   = (const float*)d_in[2];
  const float* n1b   = (const float*)d_in[3];
  const float* qkvw  = (const float*)d_in[4];
  const float* qkvb  = (const float*)d_in[5];
  const float* projw = (const float*)d_in[6];
  const float* projb = (const float*)d_in[7];
  const float* cpbw1 = (const float*)d_in[8];
  const float* cpbb1 = (const float*)d_in[9];
  const float* cpbw2 = (const float*)d_in[10];
  const float* n2g   = (const float*)d_in[11];
  const float* n2b   = (const float*)d_in[12];
  const float* w1    = (const float*)d_in[13];
  const float* b1    = (const float*)d_in[14];
  const float* w2    = (const float*)d_in[15];
  const float* b2    = (const float*)d_in[16];
  float* out = (float*)d_out;

  char* ws = (char*)d_ws;
  size_t off = 0;
  auto take = [&](size_t bytes) -> void* {
    void* p = ws + off; off += (bytes + 255) & ~(size_t)255; return p;
  };
  const size_t TOK = (size_t)32 * 64 * 64;  // 131072 tokens
  h16* wq = (h16*)take(TOK * 512 * 2);      // [win][head][tok][32]
  h16* wk = (h16*)take(TOK * 512 * 2);      // [win][head][tok][32]
  h16* wv = (h16*)take(TOK * 512 * 2);      // [win][head][32][tok]  (transposed)
  h16* hw = (h16*)take(TOK * 512 * 2);      // LN1 windows; reused as attention output
  h16* qkvw_t = (h16*)take((size_t)512 * 1536 * 2);   // [1536][512]
  h16* projw_t = (h16*)take((size_t)512 * 512 * 2);   // [512][512]
  h16* w1_t = (h16*)take((size_t)512 * 1024 * 2);     // [1024][512]
  h16* w2_t = (h16*)take((size_t)1024 * 512 * 2);     // [512][1024]
  float* bias16 = (float*)take((size_t)16 * 4096 * 4);
  h16* h2  = wv;        // reuse: v dead after attention
  h16* hid = wq;        // reuse: q+k regions (268MB) dead after attention
  h16* obuf = hw;       // reuse: LN1 windows consumed by QKV GEMM

  // weight prep: f32 [K][N] -> f16 W^T [N][K]
  wt_transpose_kernel<<<dim3(48, 16), 256, 0, stream>>>(qkvw, qkvw_t, 512, 1536);
  wt_transpose_kernel<<<dim3(16, 16), 256, 0, stream>>>(projw, projw_t, 512, 512);
  wt_transpose_kernel<<<dim3(32, 16), 256, 0, stream>>>(w1, w1_t, 512, 1024);
  wt_transpose_kernel<<<dim3(16, 32), 256, 0, stream>>>(w2, w2_t, 1024, 512);
  cpb_bias_kernel<<<1, 256, 0, stream>>>(cpbw1, cpbb1, cpbw2, bias16);
  ln1_kernel<<<2048, 256, 0, stream>>>(x, n1g, n1b, hw);
  gemm_qkv_kernel<<<dim3(1024, 12), 256, 0, stream>>>(hw, qkvw_t, qkvb, wq, wk, wv);
  attn_kernel<<<16384, 256, 0, stream>>>(wq, wk, wv, bias16, maskm, obuf);
  gemm_proj_kernel<<<dim3(1024, 4), 256, 0, stream>>>(obuf, projw_t, projb, x, out);
  ln2_kernel<<<2048, 256, 0, stream>>>(out, n2g, n2b, h2);
  gemm_mlp1_kernel<<<dim3(1024, 8), 256, 0, stream>>>(h2, w1_t, b1, hid);
  gemm_mlp2_kernel<<<dim3(1024, 4), 256, 0, stream>>>(hid, w2_t, b2, out);
  (void)in_sizes; (void)n_in; (void)out_size; (void)ws_size;
}
